// Merge_MixtralSparseMoeBlock_9637906612739
// MI455X (gfx1250) — matrix operand order repaired and checked by reading, not compiled, not run on hardware
//
#include <hip/hip_runtime.h>

// ---------------- problem constants ----------------
#define T_TOK 4096   // 4*1024 tokens
#define HDIM  4096
#define IDIM  14336
#define RNK   159
#define RP    160    // padded rank (col 159 kept zero)
#define NEXP  8
#define ISEG  8      // I-dim segments in stage2 (grid.z)
#define ICH   32     // I chunk per inner iteration (WMMA K)

typedef __attribute__((ext_vector_type(16))) __bf16 v16bf;
typedef __attribute__((ext_vector_type(8)))  float  v8f;

union BF16x16 {
  v16bf v;
  unsigned short us[16];
  uint4 q[2];
};

// Native fp32 -> bf16 convert: let clang lower the fptrunc to the gfx1250
// hardware convert (single VALU op) instead of manual bfe/add3/shift RNE.
__device__ __forceinline__ unsigned short f2bf(float f) {
  union { __bf16 h; unsigned short u; } c;
  c.h = (__bf16)f;
  return c.u;
}

__device__ __forceinline__ v8f zero8() {
  v8f r;
#pragma unroll
  for (int i = 0; i < 8; ++i) r[i] = 0.f;
  return r;
}

__device__ __forceinline__ v8f wmma_bf16(v16bf a, v16bf b, v8f c) {
  return __builtin_amdgcn_wmma_f32_16x16x32_bf16(false, a, false, b, (short)0, c,
                                                 false, false);
}

// ---- A fragment (16x32 bf16): row = lane&15, k chunks at (lane>>4)*8 and +16 ----
__device__ __forceinline__ v16bf frag_a_lds(const unsigned short* p, int stride) {
  int lane = threadIdx.x & 31;
  const unsigned short* q = p + (lane & 15) * stride + ((lane >> 4) * 8);
  BF16x16 o;
  o.q[0] = *(const uint4*)(q);
  o.q[1] = *(const uint4*)(q + 16);
  return o.v;
}

// A fragment from global fp32 (fully in-bounds, 16B aligned: stride mult of 8 floats)
__device__ __forceinline__ v16bf frag_a_f32v(const float* p, int stride) {
  int lane = threadIdx.x & 31;
  const float* q = p + (long)(lane & 15) * stride + ((lane >> 4) * 8);
  float4 a0 = *(const float4*)(q);
  float4 a1 = *(const float4*)(q + 4);
  float4 b0 = *(const float4*)(q + 16);
  float4 b1 = *(const float4*)(q + 20);
  BF16x16 o;
  o.us[0] = f2bf(a0.x); o.us[1] = f2bf(a0.y); o.us[2] = f2bf(a0.z); o.us[3] = f2bf(a0.w);
  o.us[4] = f2bf(a1.x); o.us[5] = f2bf(a1.y); o.us[6] = f2bf(a1.z); o.us[7] = f2bf(a1.w);
  o.us[8]  = f2bf(b0.x); o.us[9]  = f2bf(b0.y); o.us[10] = f2bf(b0.z); o.us[11] = f2bf(b0.w);
  o.us[12] = f2bf(b1.x); o.us[13] = f2bf(b1.y); o.us[14] = f2bf(b1.z); o.us[15] = f2bf(b1.w);
  return o.v;
}

// ---- B fragment (32x16 bf16): col n = lane&15, k chunks at (lane>>4)*8 and +16 ----
__device__ __forceinline__ v16bf frag_b_lds(const unsigned short* p, int stride) {
  int lane = threadIdx.x & 31;
  const unsigned short* q = p + (lane & 15) * stride + ((lane >> 4) * 8);
  BF16x16 o;
  o.q[0] = *(const uint4*)(q);
  o.q[1] = *(const uint4*)(q + 16);
  return o.v;
}

// B fragment from global fp32 row-major W[n][k], guarded: n<rv rows, k<cv cols.
__device__ __forceinline__ v16bf frag_b_f32s(const float* p, long stride, int rv, int cv) {
  int lane = threadIdx.x & 31;
  int n = lane & 15;
  int klo = (lane >> 4) * 8;
  BF16x16 o;
#pragma unroll
  for (int j = 0; j < 16; ++j) {
    int k = klo + j + (j >= 8 ? 8 : 0);
    float v = (n < rv && k < cv) ? p[(long)n * stride + k] : 0.f;
    o.us[j] = f2bf(v);
  }
  return o.v;
}

// =====================================================================
// K1: router logits  (T x 8) = x @ gate_w^T
// =====================================================================
__global__ __launch_bounds__(256) void k_router(const float* __restrict__ x,
                                                const float* __restrict__ gw,
                                                float* __restrict__ logits,
                                                float* __restrict__ out_logits) {
  int tid = threadIdx.x;
  int tok = blockIdx.x * 32 + (tid >> 3);
  int e = tid & 7;
  const float* xr = x + (long)tok * HDIM;
  const float* gr = gw + (long)e * HDIM;
  float acc = 0.f;
  for (int k = 0; k < HDIM; k += 4) {
    float4 a = *(const float4*)(xr + k);
    float4 b = *(const float4*)(gr + k);
    acc += a.x * b.x + a.y * b.y + a.z * b.z + a.w * b.w;
  }
  logits[tok * NEXP + e] = acc;
  out_logits[tok * NEXP + e] = acc;
}

// =====================================================================
// K2: softmax + top-2 + renormalize -> dense weight table wfull[T][8]
// =====================================================================
__global__ __launch_bounds__(256) void k_topk(const float* __restrict__ logits,
                                              float* __restrict__ wfull) {
  int t = blockIdx.x * 256 + threadIdx.x;
  if (t >= T_TOK) return;
  float l[NEXP], p[NEXP];
  float m = -1e30f;
#pragma unroll
  for (int e = 0; e < NEXP; ++e) { l[e] = logits[t * NEXP + e]; m = fmaxf(m, l[e]); }
  float s = 0.f;
#pragma unroll
  for (int e = 0; e < NEXP; ++e) { p[e] = __expf(l[e] - m); s += p[e]; }
#pragma unroll
  for (int e = 0; e < NEXP; ++e) p[e] /= s;
  int i1 = 0;
#pragma unroll
  for (int e = 1; e < NEXP; ++e) if (p[e] > p[i1]) i1 = e;
  int i2 = (i1 == 0) ? 1 : 0;
#pragma unroll
  for (int e = 0; e < NEXP; ++e) if (e != i1 && p[e] > p[i2]) i2 = e;
  float inv = 1.f / (p[i1] + p[i2]);
#pragma unroll
  for (int e = 0; e < NEXP; ++e)
    wfull[t * NEXP + e] = (e == i1) ? p[i1] * inv : ((e == i2) ? p[i2] * inv : 0.f);
}

// =====================================================================
// K0: zero a float4 region (Q accumulators)
// =====================================================================
__global__ __launch_bounds__(256) void k_zero(float4* __restrict__ p, int n4) {
  int i = blockIdx.x * 256 + threadIdx.x;
  if (i < n4) p[i] = make_float4(0.f, 0.f, 0.f, 0.f);
}

// =====================================================================
// K3 (stage1): P[z] = x @ Wv^T ; z = e*4 + chain, chains {w1v, v1, w3v, v3}
// block: 256 threads, tile [64 tokens x 160 rank]; K loop over HDIM in 32s.
// =====================================================================
__global__ __launch_bounds__(256) void k_stage1(const float* __restrict__ x,
                                                const float* __restrict__ w1v,
                                                const float* __restrict__ v1,
                                                const float* __restrict__ w3v,
                                                const float* __restrict__ v3,
                                                float* __restrict__ P) {
  __shared__ __align__(16) unsigned short As[64 * 32];   // x tile, bf16
  __shared__ __align__(16) unsigned short Bs[160 * 32];  // weight tile, bf16 (rows>=159 zero)
  int z = blockIdx.y;
  int e = z >> 2, c = z & 3;
  const float* Wb = (c == 0) ? w1v : (c == 1) ? v1 : (c == 2) ? w3v : v3;
  const float* W = Wb + (long)e * RNK * HDIM;
  float* Pout = P + (long)z * T_TOK * RP;
  int t0 = blockIdx.x * 64;
  int tid = threadIdx.x, lane = tid & 31, wave = tid >> 5;
  int mt = wave & 3;          // M tile (16 tokens)
  int nh = wave >> 2;         // this wave owns N tiles nh*5 .. nh*5+4
  v8f acc[5];
#pragma unroll
  for (int t = 0; t < 5; ++t) acc[t] = zero8();

  for (int kk = 0; kk < HDIM; kk += 32) {
    __syncthreads();
#pragma unroll
    for (int j = 0; j < 2; ++j) {  // A: 64x32 fp32 -> bf16 LDS (512 float4)
      int lin = tid + j * 256;
      int row = lin >> 3, c4 = (lin & 7) * 4;
      float4 v = *(const float4*)(x + (long)(t0 + row) * HDIM + kk + c4);
      unsigned short* d = As + row * 32 + c4;
      d[0] = f2bf(v.x); d[1] = f2bf(v.y); d[2] = f2bf(v.z); d[3] = f2bf(v.w);
    }
#pragma unroll
    for (int j = 0; j < 5; ++j) {  // B: 160x32 (rows >=159 zeroed)
      int lin = tid + j * 256;
      int row = lin >> 3, c4 = (lin & 7) * 4;
      unsigned short* d = Bs + row * 32 + c4;
      if (row < RNK) {
        float4 v = *(const float4*)(W + (long)row * HDIM + kk + c4);
        d[0] = f2bf(v.x); d[1] = f2bf(v.y); d[2] = f2bf(v.z); d[3] = f2bf(v.w);
      } else {
        d[0] = 0; d[1] = 0; d[2] = 0; d[3] = 0;
      }
    }
    __syncthreads();
    v16bf a = frag_a_lds(As + mt * 16 * 32, 32);
#pragma unroll
    for (int t = 0; t < 5; ++t) {
      v16bf b = frag_b_lds(Bs + (nh * 5 + t) * 16 * 32, 32);
      acc[t] = wmma_bf16(a, b, acc[t]);
    }
  }
  int hh = (lane >> 4) * 8, col0 = lane & 15;
#pragma unroll
  for (int t = 0; t < 5; ++t)
#pragma unroll
    for (int i = 0; i < 8; ++i)
      Pout[(long)(t0 + mt * 16 + i + hh) * RP + (nh * 5 + t) * 16 + col0] = acc[t][i];
}

// =====================================================================
// K4 (stage2): per (token-tile, expert, I-segment):
//   gate = P1@w1u^T + Pd1@u1^T ; up = P3@w3u^T + Pd3@u3^T ; h = silu(gate)*up
//   Q2  += h@w2v^T ; Qd2 += h@v2^T   (atomic add partials over segments)
// =====================================================================
__global__ __launch_bounds__(256) void k_stage2(const float* __restrict__ P,
                                                const float* __restrict__ w1u,
                                                const float* __restrict__ u1,
                                                const float* __restrict__ w3u,
                                                const float* __restrict__ u3,
                                                const float* __restrict__ w2v,
                                                const float* __restrict__ v2,
                                                float* __restrict__ Q) {
  extern __shared__ unsigned short smem2[];
  unsigned short* Ps = smem2;                     // [4][64][160] bf16 (80 KB)
  unsigned short* Hs = smem2 + 4 * 64 * RP;       // [64][32] bf16 (4 KB)
  int e = blockIdx.y, seg = blockIdx.z;
  int t0 = blockIdx.x * 64;
  int tid = threadIdx.x, lane = tid & 31, wave = tid >> 5;
  int mt = wave & 3;   // token subtile
  int nt = wave >> 2;  // 0/1: i-halves of the 32-chunk; also Q n-half (ntiles nt*5..+4)
  int hh = (lane >> 4) * 8;

  // stage the four P tiles [64 x 160] fp32 -> bf16 LDS (one-time)
#pragma unroll
  for (int j = 0; j < 40; ++j) {
    int lin = tid + j * 256;          // 10240 float4 total
    int cidx = lin / 2560;
    int rem = lin - cidx * 2560;
    int row = rem / 40, c4 = (rem - row * 40) * 4;
    const float* src = P + ((long)(e * 4 + cidx) * T_TOK + t0 + row) * RP + c4;
    float4 v = *(const float4*)src;
    unsigned short* d = Ps + (cidx * 64 + row) * RP + c4;
    d[0] = f2bf(v.x); d[1] = f2bf(v.y); d[2] = f2bf(v.z); d[3] = f2bf(v.w);
  }
  __syncthreads();

  const float* w1u_e = w1u + (long)e * IDIM * RNK;
  const float* u1_e  = u1  + (long)e * IDIM * RNK;
  const float* w3u_e = w3u + (long)e * IDIM * RNK;
  const float* u3_e  = u3  + (long)e * IDIM * RNK;
  const float* w2v_e = w2v + (long)e * RNK * IDIM;
  const float* v2_e  = v2  + (long)e * RNK * IDIM;

  v8f q2[5], qd2[5];
#pragma unroll
  for (int t = 0; t < 5; ++t) { q2[t] = zero8(); qd2[t] = zero8(); }

  const int i0 = seg * (IDIM / ISEG);
  const int i1e = i0 + (IDIM / ISEG);
  for (int ic = i0; ic < i1e; ic += ICH) {
    // hint the streaming weight rows (lowered to global_prefetch_b8)
    __builtin_prefetch(w1u_e + (long)(ic + ICH) * RNK, 0, 1);
    __builtin_prefetch(w3u_e + (long)(ic + ICH) * RNK, 0, 1);

    v8f g = zero8(), up = zero8();
    long wrow = (long)(ic + nt * 16) * RNK;
#pragma unroll
    for (int ks = 0; ks < 5; ++ks) {
      int kb = ks * 32;
      int cv = RNK - kb;
      v16bf a, b;
      a = frag_a_lds(Ps + (0 * 64 + mt * 16) * RP + kb, RP);
      b = frag_b_f32s(w1u_e + wrow + kb, RNK, 16, cv);
      g = wmma_bf16(a, b, g);
      a = frag_a_lds(Ps + (1 * 64 + mt * 16) * RP + kb, RP);
      b = frag_b_f32s(u1_e + wrow + kb, RNK, 16, cv);
      g = wmma_bf16(a, b, g);
      a = frag_a_lds(Ps + (2 * 64 + mt * 16) * RP + kb, RP);
      b = frag_b_f32s(w3u_e + wrow + kb, RNK, 16, cv);
      up = wmma_bf16(a, b, up);
      a = frag_a_lds(Ps + (3 * 64 + mt * 16) * RP + kb, RP);
      b = frag_b_f32s(u3_e + wrow + kb, RNK, 16, cv);
      up = wmma_bf16(a, b, up);
    }
    // h = silu(gate) * up -> LDS bf16
#pragma unroll
    for (int i = 0; i < 8; ++i) {
      float gv = g[i];
      float s = gv / (1.f + __expf(-gv));
      int row = mt * 16 + i + hh;
      int col = nt * 16 + (lane & 15);
      Hs[row * 32 + col] = f2bf(s * up[i]);
    }
    __syncthreads();
    // Q accumulation: K = 32 (this I chunk), N = rank
    v16bf ah = frag_a_lds(Hs + mt * 16 * 32, 32);
#pragma unroll
    for (int t = 0; t < 5; ++t) {
      int r0 = (nt * 5 + t) * 16;
      int rv = RNK - r0; if (rv > 16) rv = 16;
      v16bf bw = frag_b_f32s(w2v_e + (long)r0 * IDIM + ic, IDIM, rv, 32);
      q2[t] = wmma_bf16(ah, bw, q2[t]);
      v16bf bv = frag_b_f32s(v2_e + (long)r0 * IDIM + ic, IDIM, rv, 32);
      qd2[t] = wmma_bf16(ah, bv, qd2[t]);
    }
    __syncthreads();
  }

  float* Q2  = Q + (long)e * T_TOK * RP;
  float* Qd2 = Q + (long)(NEXP + e) * T_TOK * RP;
#pragma unroll
  for (int t = 0; t < 5; ++t)
#pragma unroll
    for (int i = 0; i < 8; ++i) {
      long row = t0 + mt * 16 + i + hh;
      int col = (nt * 5 + t) * 16 + (lane & 15);
      atomicAdd(&Q2[row * RP + col], q2[t][i]);
      atomicAdd(&Qd2[row * RP + col], qd2[t][i]);
    }
}

// =====================================================================
// K5 (stage3): final[t,h] = sum_e w_e[t] * (Q2@w2u^T + Qd2@u2^T)[t,h]
// block tile [64 tokens x 32 H-cols]; uniform per-tile expert-skip mask.
// =====================================================================
__global__ __launch_bounds__(256) void k_stage3(const float* __restrict__ Q,
                                                const float* __restrict__ w2u,
                                                const float* __restrict__ u2,
                                                const float* __restrict__ wfull,
                                                float* __restrict__ out) {
  __shared__ __align__(16) float wf[64 * NEXP];
  __shared__ int actmask;
  int t0 = blockIdx.x * 64;
  int hb = blockIdx.y * 32;
  int tid = threadIdx.x, lane = tid & 31, wave = tid >> 5;
  int mt = wave & 3, nt2 = wave >> 2;
  int hh = (lane >> 4) * 8;

  if (tid == 0) actmask = 0;
  __syncthreads();
  if (tid < 64) {
    int bits = 0;
#pragma unroll
    for (int e = 0; e < NEXP; ++e) {
      float w = wfull[(long)(t0 + tid) * NEXP + e];
      wf[tid * NEXP + e] = w;
      if (w != 0.f) bits |= (1 << e);
    }
    atomicOr(&actmask, bits);
  }
  __syncthreads();
  int mask = actmask;

  v8f f = zero8();
  for (int e = 0; e < NEXP; ++e) {
    if (!((mask >> e) & 1)) continue;
    const float* Q2  = Q + (long)e * T_TOK * RP;
    const float* Qd2 = Q + (long)(NEXP + e) * T_TOK * RP;
    const float* Wp = w2u + (long)e * HDIM * RNK;
    const float* Up = u2  + (long)e * HDIM * RNK;
    v8f ea = zero8();
    long wrow = (long)(hb + nt2 * 16) * RNK;
#pragma unroll
    for (int ks = 0; ks < 5; ++ks) {
      int kb = ks * 32;
      int cv = RNK - kb;
      v16bf a, b;
      a = frag_a_f32v(Q2 + (long)(t0 + mt * 16) * RP + kb, RP);
      b = frag_b_f32s(Wp + wrow + kb, RNK, 16, cv);
      ea = wmma_bf16(a, b, ea);
      a = frag_a_f32v(Qd2 + (long)(t0 + mt * 16) * RP + kb, RP);
      b = frag_b_f32s(Up + wrow + kb, RNK, 16, cv);
      ea = wmma_bf16(a, b, ea);
    }
#pragma unroll
    for (int i = 0; i < 8; ++i) {
      int r = mt * 16 + i + hh;
      f[i] += wf[r * NEXP + e] * ea[i];
    }
  }
#pragma unroll
  for (int i = 0; i < 8; ++i)
    out[(long)(t0 + mt * 16 + i + hh) * HDIM + hb + nt2 * 16 + (lane & 15)] = f[i];
}

// =====================================================================
// launch
// =====================================================================
extern "C" void kernel_launch(void* const* d_in, const int* in_sizes, int n_in,
                              void* d_out, int out_size, void* d_ws, size_t ws_size,
                              hipStream_t stream) {
  const float* x    = (const float*)d_in[0];
  const float* gw   = (const float*)d_in[1];
  const float* w1u  = (const float*)d_in[2];
  const float* w1v  = (const float*)d_in[3];
  const float* w2u  = (const float*)d_in[4];
  const float* w2v  = (const float*)d_in[5];
  const float* w3u  = (const float*)d_in[6];
  const float* w3v  = (const float*)d_in[7];
  const float* u1   = (const float*)d_in[8];
  const float* v1   = (const float*)d_in[9];
  const float* u2   = (const float*)d_in[10];
  const float* v2   = (const float*)d_in[11];
  const float* u3   = (const float*)d_in[12];
  const float* v3   = (const float*)d_in[13];
  float* out = (float*)d_out;

  float* ws = (float*)d_ws;
  float* ws_logits = ws;                                    // T*8
  float* ws_w = ws_logits + (size_t)T_TOK * NEXP;           // T*8
  float* ws_P = ws_w + (size_t)T_TOK * NEXP;                // 32 * T * 160
  float* ws_Q = ws_P + (size_t)32 * T_TOK * RP;             // 16 * T * 160

  // K1: router logits (also writes 2nd output: d_out[T*H ..])
  k_router<<<dim3(T_TOK / 32), 256, 0, stream>>>(x, gw, ws_logits,
                                                 out + (size_t)T_TOK * HDIM);
  // K2: top-2 softmax weights
  k_topk<<<dim3(T_TOK / 256), 256, 0, stream>>>(ws_logits, ws_w);
  // K0: zero Q accumulators
  {
    int n4 = (16 * T_TOK * RP) / 4;
    k_zero<<<dim3((n4 + 255) / 256), 256, 0, stream>>>((float4*)ws_Q, n4);
  }
  // K3: stage1 rank projections (32 z-slices: expert x {w1v,v1,w3v,v3})
  k_stage1<<<dim3(T_TOK / 64, 32), 256, 0, stream>>>(x, w1v, v1, w3v, v3, ws_P);
  // K4: stage2 fused gate/up/silu + rank back-projection (dynamic LDS: 84 KB)
  {
    size_t smem = (size_t)(4 * 64 * RP + 64 * 32) * sizeof(unsigned short);
    k_stage2<<<dim3(T_TOK / 64, NEXP, ISEG), 256, smem, stream>>>(
        ws_P, w1u, u1, w3u, u3, w2v, v2, ws_Q);
  }
  // K5: stage3 down-projection + routed combine
  k_stage3<<<dim3(T_TOK / 64, HDIM / 32), 256, 0, stream>>>(ws_Q, w2u, u2, ws_w, out);
}
